// ConvFormer_11355893531224
// MI455X (gfx1250) — matrix operand order repaired and checked by reading, not compiled, not run on hardware
//
#include <hip/hip_runtime.h>
#include <cstdint>

#define BB 8
#define LL 4096
#define DD 512
#define BL (BB * LL)

typedef __attribute__((ext_vector_type(16))) __bf16   bf16x16;
typedef __attribute__((ext_vector_type(8)))  float    f32x8;
typedef __attribute__((ext_vector_type(8)))  uint32_t u32x8;

// ---- packed bf16 conversion helpers (RNE via bit ops) ----
__device__ __forceinline__ uint32_t rne16(float f) {
  union { float f; uint32_t u; } c; c.f = f;
  return (c.u + 0x7FFFu + ((c.u >> 16) & 1u)) >> 16;
}
__device__ __forceinline__ uint32_t pack2(float lo, float hi) {
  return rne16(lo) | (rne16(hi) << 16);
}
__device__ __forceinline__ void store_val(float* p, float v)  { *p = v; }
__device__ __forceinline__ void store_val(__bf16* p, float v) {
  union { uint16_t s; __bf16 b; } r; r.s = (uint16_t)rne16(v); *p = r.b;
}

// copy 4 packed words (16B) of a 16-word row; q in [0,4)
__device__ __forceinline__ void stage4(uint32_t* d, const uint32_t* s, int q) {
  reinterpret_cast<uint4*>(d)[q] = reinterpret_cast<const uint4*>(s)[q];
}

// conv epilogue: fp32 (last conv) or packed bf16 (chained convs)
__device__ __forceinline__ void conv_store_row(float* out, size_t row, int obase, int hi,
                                               const f32x8& a, const float* bv) {
  float* p = out + row * DD + obase + 8 * hi;
  #pragma unroll
  for (int r = 0; r < 8; ++r) p[r] = a[r] + bv[r];
}
__device__ __forceinline__ void conv_store_row(uint32_t* out, size_t row, int obase, int hi,
                                               const f32x8& a, const float* bv) {
  uint32_t* p = out + row * (DD / 2) + (obase >> 1) + 4 * hi;
  #pragma unroll
  for (int q = 0; q < 4; ++q)
    p[q] = pack2(a[2 * q] + bv[2 * q], a[2 * q + 1] + bv[2 * q + 1]);
}

// =====================================================================
// One-shot pre-pass kernels
// =====================================================================
__global__ __launch_bounds__(256)
void pack_rows_kernel(const float* __restrict__ src, uint32_t* __restrict__ dst)
{
  int id = blockIdx.x * 256 + threadIdx.x;
  float4 v = reinterpret_cast<const float4*>(src)[id];
  uint2 w; w.x = pack2(v.x, v.y); w.y = pack2(v.z, v.w);
  reinterpret_cast<uint2*>(dst)[id] = w;
}

// W (D,D,K) fp32 -> dst[(tap*D + o)*(D/2) + cw] = pair(W[o,2cw,tap], W[o,2cw+1,tap])
template<int K>
__global__ __launch_bounds__(256)
void pack_convw_kernel(const float* __restrict__ W, uint32_t* __restrict__ dst)
{
  int id = blockIdx.x * 256 + threadIdx.x;        // (tap*512 + o)*256 + cw
  int cw = id & 255;
  int o  = (id >> 8) & 511;
  int tap = id >> 17;
  if (tap >= K) return;
  const float* p = W + ((size_t)o * DD + 2 * cw) * K + tap;
  dst[id] = pack2(p[0], p[K]);
}

// src (Kdim, N) fp32 -> dst[n*(Kdim/2) + kw] = pair(src[2kw,n], src[2kw+1,n])
__global__ __launch_bounds__(256)
void transpose_pack_kernel(const float* __restrict__ src, uint32_t* __restrict__ dst,
                           int Kdim, int N)
{
  int id = blockIdx.x * 256 + threadIdx.x;
  int Kw = Kdim >> 1;
  if (id >= N * Kw) return;
  int n = id % N, kw = id / N;
  dst[(size_t)n * Kw + kw] = pack2(src[(size_t)(2 * kw) * N + n],
                                   src[(size_t)(2 * kw + 1) * N + n]);
}

// =====================================================================
// Masked segment conv (input always packed bf16 words [row][D/2]):
//   out[b,l,o] = bias[o] + sum_{tap,c} W[o,c,tap]*x[b,l+tap-PAD,c]*mask
// Chain masks hoisted to per-tile bitmasks; inner loop is pure
// b128 staging + b128 fragment loads + v_and mask + wmma.
// =====================================================================
template<int K, typename TOut>
__global__ __launch_bounds__(256)
void conv_masked_kernel(const uint32_t* __restrict__ xin,  // packed (B,L,D/2)
                        const int* __restrict__ chain,     // (B,L)
                        const uint32_t* __restrict__ wpk,  // [tap][o][cw] packed
                        const float* __restrict__ bias,    // (D)
                        TOut* __restrict__ out)            // f32 (B,L,D) or packed
{
  constexpr int PAD  = (K - 1) / 2;
  constexpr int LBLK = 128;
  constexpr int NL   = LBLK + K - 1;

  __shared__ uint32_t sW32[128 * 16];
  __shared__ uint32_t sX32[NL * 16];
  __shared__ int      sCh[NL];

  const int tid  = threadIdx.x;
  const int wave = tid >> 5;
  const int lane = tid & 31;
  const int m    = lane & 15;
  const int hi   = lane >> 4;

  const int nblk_l = LL / LBLK;
  const int b  = blockIdx.x / nblk_l;
  const int l0 = (blockIdx.x % nblk_l) * LBLK;
  const int o0 = blockIdx.y * 128;

  if (tid < NL) {
    int gl = l0 - PAD + tid;
    sCh[tid] = (gl >= 0 && gl < LL) ? chain[b * LL + gl] : -1;
  }
  __syncthreads();

  // per-tile chain masks: chm[t] bit 'tap' = contribution enabled
  uint32_t chm[8];
  #pragma unroll
  for (int t = 0; t < 8; ++t) {
    int cen = PAD + t * 16 + m;
    int cc = sCh[cen];
    uint32_t bits = 0u;
    #pragma unroll
    for (int tap = 0; tap < K; ++tap)
      bits |= (sCh[cen + tap - PAD] == cc) ? (1u << tap) : 0u;
    chm[t] = bits;
  }

  f32x8 acc[8];
  #pragma unroll
  for (int t = 0; t < 8; ++t)
    #pragma unroll
    for (int r = 0; r < 8; ++r) acc[t][r] = 0.f;

  for (int c0 = 0; c0 < DD; c0 += 32) {
    __syncthreads();
    // stage x window: two exact passes + (K-1)*4 tail threads; clamped rows
    #pragma unroll
    for (int pass = 0; pass < 2; ++pass) {
      int i = tid + pass * 256;
      int ll = i >> 2, q = i & 3;
      int gl = min(max(l0 - PAD + ll, 0), LL - 1);
      stage4(&sX32[ll * 16], xin + ((size_t)b * LL + gl) * (DD / 2) + (c0 >> 1), q);
    }
    if (tid < (K - 1) * 4) {
      int i = tid + 512;
      int ll = i >> 2, q = i & 3;
      int gl = min(l0 - PAD + ll, LL - 1);    // ll >= 128, no low clamp needed
      stage4(&sX32[ll * 16], xin + ((size_t)b * LL + gl) * (DD / 2) + (c0 >> 1), q);
    }
    for (int tap = 0; tap < K; ++tap) {
      if (tap) __syncthreads();
      // stage pre-packed W tile: two exact b128-copy passes
      #pragma unroll
      for (int pass = 0; pass < 2; ++pass) {
        int i = tid + pass * 256;
        int ol = i >> 2, q = i & 3;
        stage4(&sW32[ol * 16],
               wpk + ((size_t)tap * DD + o0 + ol) * (DD / 2) + (c0 >> 1), q);
      }
      __syncthreads();

      // A fragment (16 o x 32 c)
      u32x8 au;
      const uint32_t* aw = &sW32[(wave * 16 + m) * 16];
      #pragma unroll
      for (int j = 0; j < 8; ++j)
        au[j] = aw[(hi << 3) + j];
      bf16x16 af = __builtin_bit_cast(bf16x16, au);

      const int d = tap - PAD;
      #pragma unroll
      for (int t = 0; t < 8; ++t) {
        int pl = PAD + t * 16 + m + d;
        uint32_t msk = 0u - ((chm[t] >> tap) & 1u);   // all-ones or zero
        const uint32_t* bx = &sX32[pl * 16 + (hi << 3)];
        u32x8 bu;
        #pragma unroll
        for (int j = 0; j < 8; ++j)
          bu[j] = bx[j] & msk;
        bf16x16 bfrag = __builtin_bit_cast(bf16x16, bu);

        acc[t] = __builtin_amdgcn_wmma_f32_16x16x32_bf16(
            false, af, false, bfrag, (short)0, acc[t], false, false);
      }
    }
  }

  const int obase = o0 + wave * 16;
  float bv[8];
  #pragma unroll
  for (int r = 0; r < 8; ++r) bv[r] = bias[obase + r + 8 * hi];
  #pragma unroll
  for (int t = 0; t < 8; ++t)
    conv_store_row(out, (size_t)b * LL + l0 + t * 16 + m, obase, hi, acc[t], bv);
}

// =====================================================================
// C[M,N] = act( A[M,K] @ B[K,N] + bias[N] )
// A packed bf16 words [m][K/2]; B pre-packed [n][K/2]. Pure-copy staging.
// =====================================================================
template<bool GELU, typename TOut>
__global__ __launch_bounds__(256)
void gemm_bias_kernel(const uint32_t* __restrict__ A, const uint32_t* __restrict__ Bt,
                      const float* __restrict__ bias, TOut* __restrict__ Cout,
                      int Kdim, int N)
{
  __shared__ uint32_t sA32[128 * 16];
  __shared__ uint32_t sB32[128 * 16];

  const int tid = threadIdx.x, wave = tid >> 5, lane = tid & 31;
  const int m = lane & 15, hi = lane >> 4;
  const int m0 = blockIdx.x * 128, n0 = blockIdx.y * 128;
  const int Kw = Kdim >> 1;

  f32x8 acc[8];
  #pragma unroll
  for (int t = 0; t < 8; ++t)
    #pragma unroll
    for (int r = 0; r < 8; ++r) acc[t][r] = 0.f;

  for (int k0 = 0; k0 < Kdim; k0 += 32) {
    __syncthreads();
    #pragma unroll
    for (int pass = 0; pass < 2; ++pass) {
      int i = tid + pass * 256;
      int ml = i >> 2, q = i & 3;
      stage4(&sA32[ml * 16], A + (size_t)(m0 + ml) * Kw + (k0 >> 1), q);
    }
    #pragma unroll
    for (int pass = 0; pass < 2; ++pass) {
      int i = tid + pass * 256;
      int nl = i >> 2, q = i & 3;
      stage4(&sB32[nl * 16], Bt + (size_t)(n0 + nl) * Kw + (k0 >> 1), q);
    }
    if (k0 + 32 < Kdim)
      __builtin_prefetch(Bt + (size_t)(n0 + (tid & 127)) * Kw + ((k0 + 32) >> 1), 0, 1);
    __syncthreads();

    const uint32_t* pb = &sB32[(wave * 16 + m) * 16 + (hi << 3)];
    u32x8 bu;
    #pragma unroll
    for (int j = 0; j < 8; ++j) bu[j] = pb[j];
    bf16x16 bfrag = __builtin_bit_cast(bf16x16, bu);

    #pragma unroll
    for (int t = 0; t < 8; ++t) {
      const uint32_t* pa = &sA32[(t * 16 + m) * 16];
      u32x8 au;
      #pragma unroll
      for (int j = 0; j < 8; ++j)
        au[j] = pa[(hi << 3) + j];
      bf16x16 af = __builtin_bit_cast(bf16x16, au);

      acc[t] = __builtin_amdgcn_wmma_f32_16x16x32_bf16(
          false, af, false, bfrag, (short)0, acc[t], false, false);
    }
  }

  const int n = n0 + wave * 16 + m;
  const float bn = bias[n];
  #pragma unroll
  for (int t = 0; t < 8; ++t) {
    #pragma unroll
    for (int r = 0; r < 8; ++r) {
      float v = acc[t][r] + bn;
      if (GELU) {
        float x3 = v * v * v;
        v = 0.5f * v * (1.f + tanhf(0.7978845608028654f * (v + 0.044715f * x3)));
      }
      size_t mm = (size_t)(m0 + t * 16 + r + 8 * hi);
      store_val(&Cout[mm * (size_t)N + n], v);
    }
  }
}

// =====================================================================
// out = LN(a + c)*g + beta; lane owns 16 contiguous channels so an
// optional packed-bf16 copy can be emitted with in-lane pair packing.
// =====================================================================
__global__ __launch_bounds__(256)
void ln_residual_kernel(const float* __restrict__ a, const float* __restrict__ c,
                        const float* __restrict__ g, const float* __restrict__ beta,
                        float* __restrict__ out, uint32_t* __restrict__ packed_out)
{
  const int wave = threadIdx.x >> 5, lane = threadIdx.x & 31;
  const size_t row = (size_t)blockIdx.x * 8 + wave;
  const float* pa = a + row * DD + lane * 16;
  const float* pc = c + row * DD + lane * 16;
  float v[16], s = 0.f, sq = 0.f;
  #pragma unroll
  for (int i = 0; i < 4; ++i) {
    float4 xa = reinterpret_cast<const float4*>(pa)[i];
    float4 xc = reinterpret_cast<const float4*>(pc)[i];
    float t0 = xa.x + xc.x, t1 = xa.y + xc.y, t2 = xa.z + xc.z, t3 = xa.w + xc.w;
    v[4 * i] = t0; v[4 * i + 1] = t1; v[4 * i + 2] = t2; v[4 * i + 3] = t3;
    s += t0 + t1 + t2 + t3;
    sq += t0 * t0 + t1 * t1 + t2 * t2 + t3 * t3;
  }
  #pragma unroll
  for (int off = 16; off > 0; off >>= 1) {
    s  += __shfl_xor(s,  off, 32);
    sq += __shfl_xor(sq, off, 32);
  }
  const float mean = s * (1.f / DD);
  const float var  = sq * (1.f / DD) - mean * mean;
  const float rs   = rsqrtf(var + 1e-5f);

  const float* pg = g + lane * 16;
  const float* pb = beta + lane * 16;
  float o[16];
  #pragma unroll
  for (int i = 0; i < 16; ++i)
    o[i] = (v[i] - mean) * rs * pg[i] + pb[i];

  float* po = out + row * DD + lane * 16;
  #pragma unroll
  for (int i = 0; i < 4; ++i) {
    float4 w; w.x = o[4 * i]; w.y = o[4 * i + 1]; w.z = o[4 * i + 2]; w.w = o[4 * i + 3];
    reinterpret_cast<float4*>(po)[i] = w;
  }
  if (packed_out) {
    uint32_t* pp = packed_out + row * (DD / 2) + lane * 8;
    #pragma unroll
    for (int i = 0; i < 2; ++i) {
      uint4 w;
      w.x = pack2(o[8 * i],     o[8 * i + 1]);
      w.y = pack2(o[8 * i + 2], o[8 * i + 3]);
      w.z = pack2(o[8 * i + 4], o[8 * i + 5]);
      w.w = pack2(o[8 * i + 6], o[8 * i + 7]);
      reinterpret_cast<uint4*>(pp)[i] = w;
    }
  }
}

// =====================================================================
extern "C" void kernel_launch(void* const* d_in, const int* in_sizes, int n_in,
                              void* d_out, int out_size, void* d_ws, size_t ws_size,
                              hipStream_t stream)
{
  const float* x     = (const float*)d_in[0];
  const int*   chain = (const int*)  d_in[1];
  const float* W3    = (const float*)d_in[2];
  const float* b3    = (const float*)d_in[3];
  const float* W5    = (const float*)d_in[4];
  const float* b5    = (const float*)d_in[5];
  const float* W7    = (const float*)d_in[6];
  const float* b7    = (const float*)d_in[7];
  const float* mw1   = (const float*)d_in[8];
  const float* mb1   = (const float*)d_in[9];
  const float* mw2   = (const float*)d_in[10];
  const float* mb2   = (const float*)d_in[11];
  const float* g1    = (const float*)d_in[12];
  const float* be1   = (const float*)d_in[13];
  const float* g2    = (const float*)d_in[14];
  const float* be2   = (const float*)d_in[15];
  float* out = (float*)d_out;

  // ---- workspace layout (~301 MB) ----
  char* ws = (char*)d_ws;
  const size_t MB = 1024ull * 1024ull;
  float*    buf0 = (float*)ws;                   //  64 MB: conv3 out / mlp out (f32)
  float*    h    = (float*)(ws + 64 * MB);       //  64 MB: LN1 output (f32)
  char*     actr = ws + 128 * MB;                // 128 MB: act bf16 region
  __bf16*   act  = (__bf16*)actr;                //   (BL x 2048 bf16)
  // overlaid on act region (all dead before GEMM1 writes act):
  uint32_t* cp0  = (uint32_t*)actr;              //  32 MB conv packed ping
  uint32_t* cp1  = (uint32_t*)(actr + 32 * MB);  //  32 MB conv packed pong
  uint32_t* xbf  = (uint32_t*)(actr + 64 * MB);  //  32 MB packed x
  uint32_t* hbf  = (uint32_t*)(ws + 256 * MB);   //  32 MB packed h
  uint32_t* wc3  = (uint32_t*)(ws + 288 * MB);   // 1.5 MB
  uint32_t* wc5  = (uint32_t*)(ws + 290 * MB);   // 2.5 MB
  uint32_t* wc7  = (uint32_t*)(ws + 293 * MB);   // 3.5 MB
  uint32_t* w1t  = (uint32_t*)(ws + 297 * MB);   // 2.0 MB (2048 x 256 w)
  uint32_t* w2t  = (uint32_t*)(ws + 299 * MB);   // 2.0 MB (512 x 1024 w)

  dim3 blk(256);

  // ---- one-shot packing ----
  pack_rows_kernel<<<BL / 2, blk, 0, stream>>>(x, xbf);
  pack_convw_kernel<3><<<3 * 512, blk, 0, stream>>>(W3, wc3);
  pack_convw_kernel<5><<<5 * 512, blk, 0, stream>>>(W5, wc5);
  pack_convw_kernel<7><<<7 * 512, blk, 0, stream>>>(W7, wc7);
  transpose_pack_kernel<<<(2048 * 256 + 255) / 256, blk, 0, stream>>>(mw1, w1t, DD, 4 * DD);
  transpose_pack_kernel<<<(512 * 1024 + 255) / 256, blk, 0, stream>>>(mw2, w2t, 4 * DD, DD);

  // ---- conv chain ----
  dim3 gConv(BL / 128, DD / 128);
  conv_masked_kernel<3, uint32_t><<<gConv, blk, 0, stream>>>(xbf, chain, wc3, b3, cp0);
  conv_masked_kernel<5, uint32_t><<<gConv, blk, 0, stream>>>(cp0, chain, wc5, b5, cp1);
  conv_masked_kernel<7, float   ><<<gConv, blk, 0, stream>>>(cp1, chain, wc7, b7, buf0);

  // h = LN1(x + conv3) (f32 + packed bf16)
  ln_residual_kernel<<<BL / 8, blk, 0, stream>>>(x, buf0, g1, be1, h, hbf);

  // act = gelu(h @ W1 + b1) as bf16
  gemm_bias_kernel<true, __bf16>
      <<<dim3(BL / 128, (4 * DD) / 128), blk, 0, stream>>>(hbf, w1t, mb1, act, DD, 4 * DD);
  // mlp = act @ W2 + b2 -> buf0 (f32)
  gemm_bias_kernel<false, float>
      <<<dim3(BL / 128, DD / 128), blk, 0, stream>>>((const uint32_t*)act, w2t, mb2,
                                                     buf0, 4 * DD, DD);

  // out = LN2(h + mlp)
  ln_residual_kernel<<<BL / 8, blk, 0, stream>>>(h, buf0, g2, be2, out, nullptr);
}
